// STMambaVisionLinear_12438225289401
// MI455X (gfx1250) — compile-verified, matrix operand order checked by reading, not run on hardware
//
#include <hip/hip_runtime.h>
#include <hip/hip_bf16.h>
#include <cstdint>

typedef __attribute__((ext_vector_type(16))) __bf16 v16bf;
typedef __attribute__((ext_vector_type(8)))  float  v8f;

#define BT_   64
#define P_    1024
#define C_    1024
#define T_    8
#define B_    8
#define RANK_ 32
#define DST_  16
#define SCAL_ 4.0f
#define MTOT_ (BT_ * P_)   // 65536 rows

// ---------- helpers ----------

// Pack two fp32 -> packed bf16 pair with a single v_perm_b32 (truncate).
__device__ __forceinline__ unsigned pack_bf16(float lo, float hi) {
  return __builtin_amdgcn_perm(__builtin_bit_cast(unsigned, hi),
                               __builtin_bit_cast(unsigned, lo),
                               0x07060302u);
}

struct Frag32B { uint4 lo, hi; };

// LDS tile rows are 16 dwords (32 bf16) each.  Fragment layout for
// v_wmma_*_16x16x32 (16-bit A/B): lane&15 = matrix row, lane>>4 selects
// K halves {0..7,16..23} (ksel=0) vs {8..15,24..31} (ksel=1).
__device__ __forceinline__ v16bf load_frag(const unsigned* base, int row, int ksel) {
  const uint4* p0 = (const uint4*)(base + row * 16 + ksel * 4);
  const uint4* p1 = (const uint4*)(base + row * 16 + 8 + ksel * 4);
  Frag32B f;
  f.lo = *p0;
  f.hi = *p1;
  return __builtin_bit_cast(v16bf, f);
}

// Convert 16 staged floats -> 8 packed dwords -> two b128 LDS stores.
__device__ __forceinline__ void cvt_store_row(unsigned* dst, const float4* v) {
  uint4 lo, hi;
  lo.x = pack_bf16(v[0].x, v[0].y); lo.y = pack_bf16(v[0].z, v[0].w);
  lo.z = pack_bf16(v[1].x, v[1].y); lo.w = pack_bf16(v[1].z, v[1].w);
  hi.x = pack_bf16(v[2].x, v[2].y); hi.y = pack_bf16(v[2].z, v[2].w);
  hi.z = pack_bf16(v[3].x, v[3].y); hi.w = pack_bf16(v[3].z, v[3].w);
  *(uint4*)(dst)     = lo;
  *(uint4*)(dst + 4) = hi;
}

__device__ __forceinline__ void load4(float4* r, const float* p) {
  const float4* g = (const float4*)p;
  r[0] = g[0]; r[1] = g[1]; r[2] = g[2]; r[3] = g[3];
}

#define WMMA_BF16(A, Bm, Cacc) \
  __builtin_amdgcn_wmma_f32_16x16x32_bf16(false, (A), false, (Bm), (short)0, (Cacc), false, false)

// ---------- K1: red = x @ Wdown^T   (65536x1024 @ 1024x32) ----------

__global__ __launch_bounds__(256) void k1_down(const float* __restrict__ x,
                                               const float* __restrict__ Wdown,
                                               float* __restrict__ red) {
  __shared__ unsigned sA[128 * 16];   // 128 rows x 32 bf16
  __shared__ unsigned sB[32 * 16];    // 32 rows x 32 bf16

  const int tid   = threadIdx.x;
  const int lane  = tid & 31;
  const int wave  = tid >> 5;
  const int m0    = blockIdx.x * 128;
  const int arow  = tid >> 1;
  const int ahalf = tid & 1;

  v8f acc0 = {};
  v8f acc1 = {};

  float4 ra[4], rb[4];
  load4(ra, x + (size_t)(m0 + arow) * C_ + ahalf * 16);
  if (tid < 64) load4(rb, Wdown + (size_t)arow * C_ + ahalf * 16);

  for (int kk = 0; kk < 32; ++kk) {
    __syncthreads();                       // previous compute done with LDS
    cvt_store_row(sA + arow * 16 + ahalf * 8, ra);
    if (tid < 64) cvt_store_row(sB + arow * 16 + ahalf * 8, rb);
    if (kk + 1 < 32) {                     // prefetch next chunk into regs
      const int k0 = (kk + 1) * 32;
      load4(ra, x + (size_t)(m0 + arow) * C_ + k0 + ahalf * 16);
      if (tid < 64) load4(rb, Wdown + (size_t)arow * C_ + k0 + ahalf * 16);
    }
    __syncthreads();                       // tile visible to all waves

    const int r    = lane & 15;
    const int ksel = lane >> 4;
    v16bf a  = load_frag(sA, wave * 16 + r, ksel);
    v16bf b0 = load_frag(sB, r, ksel);
    v16bf b1 = load_frag(sB, 16 + r, ksel);
    acc0 = WMMA_BF16(a, b0, acc0);
    acc1 = WMMA_BF16(a, b1, acc1);
  }

  const int col   = lane & 15;
  const int rbase = m0 + wave * 16 + (lane >> 4) * 8;
#pragma unroll
  for (int v = 0; v < 8; ++v) {
    red[(size_t)(rbase + v) * RANK_ + col]      = acc0[v];
    red[(size_t)(rbase + v) * RANK_ + 16 + col] = acc1[v];
  }
}

// ---------- K2: per-sequence SSM scan (one wave32 per sequence) ----------

__device__ __forceinline__ float softplus_f(float v) {
  return v > 20.0f ? v : log1pf(__expf(v));
}

__global__ __launch_bounds__(256) void k2_ssm(const float* __restrict__ red,
                                              float* __restrict__ redo,
                                              const float* __restrict__ A_log,
                                              const float* __restrict__ Dp,
                                              const float* __restrict__ Wxp,
                                              const float* __restrict__ bxp,
                                              const float* __restrict__ Wout,
                                              const float* __restrict__ bout) {
  __shared__ float sWxp[64 * 32];
  __shared__ float sWout[32 * 32];

  const int tid  = threadIdx.x;
  const int lane = tid & 31;

  for (int i = tid; i < 64 * 32; i += 256) sWxp[i] = Wxp[i];
  for (int i = tid; i < 32 * 32; i += 256) sWout[i] = Wout[i];
  __syncthreads();

  const int seq = blockIdx.x * 8 + (tid >> 5);   // 8192 sequences
  const int b   = seq >> 10;                     // / P_
  const int p   = seq & (P_ - 1);

  float Arow[DST_];
#pragma unroll
  for (int s = 0; s < DST_; ++s) Arow[s] = -__expf(A_log[lane * DST_ + s]);
  const float dp  = Dp[lane];
  const float bx1 = bxp[lane];
  const float bx2 = bxp[32 + lane];
  const float bo  = bout[lane];

  float h[DST_];
#pragma unroll
  for (int s = 0; s < DST_; ++s) h[s] = 0.0f;

  for (int t = 0; t < T_; ++t) {
    const size_t row = (size_t)((b * T_ + t) * P_ + p);
    const float xt = red[row * RANK_ + lane];

    // proj = x_t @ Wxp^T + bxp : lane computes proj[lane] and proj[32+lane]
    float a1 = bx1, a2 = bx2;
#pragma unroll
    for (int rr = 0; rr < 32; ++rr) {
      const float xs = __shfl(xt, rr, 32);
      a1 += sWxp[lane * 32 + rr] * xs;
      a2 += sWxp[(32 + lane) * 32 + rr] * xs;
    }
    const float dlt = softplus_f(a1);

    // state update + y = h.C + Dp*x   (Bm from lanes 0..15, Cm from 16..31)
    float y = dp * xt;
#pragma unroll
    for (int s = 0; s < DST_; ++s) {
      const float Bs = __shfl(a2, s, 32);
      const float Cs = __shfl(a2, 16 + s, 32);
      h[s] = __expf(dlt * Arow[s]) * h[s] + dlt * Bs * xt;
      y += h[s] * Cs;
    }

    // out_t = y @ Wout^T + bout, pre-scaled so K3 folds delta directly
    float o = bo;
#pragma unroll
    for (int rr = 0; rr < 32; ++rr) {
      const float ys = __shfl(y, rr, 32);
      o += sWout[lane * 32 + rr] * ys;
    }
    redo[row * RANK_ + lane] = SCAL_ * o;
  }
}

// ---------- K3: out = x @ Wb^T + bb + redo @ Wup^T ----------

__global__ __launch_bounds__(256) void k3_main(const float* __restrict__ x,
                                               const float* __restrict__ Wb,
                                               const float* __restrict__ bb,
                                               const float* __restrict__ Wup,
                                               const float* __restrict__ redo,
                                               float* __restrict__ out) {
  __shared__ unsigned sA[128 * 16];   // A tile: 128 x 32 bf16
  __shared__ unsigned sB[128 * 16];   // B tile: 128 x 32 bf16

  const int tid   = threadIdx.x;
  const int lane  = tid & 31;
  const int wave  = tid >> 5;
  const int wr    = wave & 3;    // 4 wave-rows -> 32 output rows each
  const int wc    = wave >> 2;   // 2 wave-cols -> 64 output cols each
  const int n0    = blockIdx.x * 128;
  const int m0    = blockIdx.y * 128;
  const int arow  = tid >> 1;
  const int ahalf = tid & 1;

  v8f acc[2][4];
#pragma unroll
  for (int i = 0; i < 2; ++i)
#pragma unroll
    for (int j = 0; j < 4; ++j) acc[i][j] = v8f{};

  float4 ra[4], rb[4];
  load4(ra, x  + (size_t)(m0 + arow) * C_ + ahalf * 16);
  load4(rb, Wb + (size_t)(n0 + arow) * C_ + ahalf * 16);

  for (int kk = 0; kk < 33; ++kk) {
    __syncthreads();                       // previous compute done with LDS
    cvt_store_row(sA + arow * 16 + ahalf * 8, ra);
    cvt_store_row(sB + arow * 16 + ahalf * 8, rb);

    if (kk + 1 < 32) {                     // prefetch next main chunk
      const int k0 = (kk + 1) * 32;
      load4(ra, x  + (size_t)(m0 + arow) * C_ + k0 + ahalf * 16);
      load4(rb, Wb + (size_t)(n0 + arow) * C_ + k0 + ahalf * 16);
    } else if (kk + 1 == 32) {             // prefetch delta chunk (pre-scaled redo)
      load4(ra, redo + (size_t)(m0 + arow) * RANK_ + ahalf * 16);
      load4(rb, Wup  + (size_t)(n0 + arow) * RANK_ + ahalf * 16);
    }
    __syncthreads();                       // tile visible to all waves

    const int r    = lane & 15;
    const int ksel = lane >> 4;
    v16bf a0 = load_frag(sA, wr * 32 + r, ksel);
    v16bf a1 = load_frag(sA, wr * 32 + 16 + r, ksel);
#pragma unroll
    for (int j = 0; j < 4; ++j) {
      v16bf bj = load_frag(sB, wc * 64 + j * 16 + r, ksel);
      acc[0][j] = WMMA_BF16(a0, bj, acc[0][j]);
      acc[1][j] = WMMA_BF16(a1, bj, acc[1][j]);
    }
  }

  // epilogue: add bb and store (each element written exactly once)
  const int col  = lane & 15;
  const int rsel = (lane >> 4) * 8;
#pragma unroll
  for (int i = 0; i < 2; ++i) {
    const int rowb = m0 + wr * 32 + i * 16 + rsel;
#pragma unroll
    for (int j = 0; j < 4; ++j) {
      const int ncol = n0 + wc * 64 + j * 16 + col;
      const float bias = bb[ncol];
#pragma unroll
      for (int v = 0; v < 8; ++v)
        out[(size_t)(rowb + v) * C_ + ncol] = acc[i][j][v] + bias;
    }
  }
}

// ---------- launch ----------

extern "C" void kernel_launch(void* const* d_in, const int* in_sizes, int n_in,
                              void* d_out, int out_size, void* d_ws, size_t ws_size,
                              hipStream_t stream) {
  (void)in_sizes; (void)n_in; (void)out_size; (void)ws_size;
  const float* x     = (const float*)d_in[0];
  const float* Wb    = (const float*)d_in[1];
  const float* bb    = (const float*)d_in[2];
  const float* Wdown = (const float*)d_in[3];
  const float* Wup   = (const float*)d_in[4];
  const float* A_log = (const float*)d_in[5];
  const float* Dp    = (const float*)d_in[6];
  const float* Wxp   = (const float*)d_in[7];
  const float* bxp   = (const float*)d_in[8];
  const float* Wout  = (const float*)d_in[9];
  const float* bout  = (const float*)d_in[10];
  float* out  = (float*)d_out;

  float* red  = (float*)d_ws;                         // 65536 x 32 = 8 MB
  float* redo = red + (size_t)MTOT_ * RANK_;          // 65536 x 32 = 8 MB

  k1_down<<<MTOT_ / 128, 256, 0, stream>>>(x, Wdown, red);
  k2_ssm<<<(B_ * P_) / 8, 256, 0, stream>>>(red, redo, A_log, Dp, Wxp, bxp, Wout, bout);
  k3_main<<<dim3(C_ / 128, MTOT_ / 128), 256, 0, stream>>>(x, Wb, bb, Wup, redo, out);
}
